// FeatureAdaption_67233418051681
// MI455X (gfx1250) — matrix operand order repaired and checked by reading, not compiled, not run on hardware
//
#include <hip/hip_runtime.h>
#include <hip/hip_bf16.h>

typedef __attribute__((ext_vector_type(16))) __bf16 v16bf;
typedef __attribute__((ext_vector_type(8)))  __bf16 v8bf;
typedef __attribute__((ext_vector_type(8)))  float  v8f;
typedef __attribute__((ext_vector_type(8)))  unsigned int v8u;
typedef __attribute__((ext_vector_type(4)))  unsigned int v4u;
typedef __attribute__((ext_vector_type(8)))  int v8i;
typedef __attribute__((ext_vector_type(4)))  int v4i;
typedef unsigned int u32;

#define H_H 126          // h spatial
#define OFF_HW 128       // offset / output spatial
#define CIN 256
#define COUT 256
#define KK 2304          // C * 9
#define PIX 64           // pixels per block
#define ASTR 40          // LDS row stride (bf16) for A tile (32 data + 8 pad)
#define BSTR 40          // LDS row stride (bf16) for B tile

__device__ __forceinline__ int clampi(int v, int lo, int hi) {
    return v < lo ? lo : (v > hi ? hi : v);
}

// ---------- ConvTranspose2d(4->18, k=4, s=1) + ReLU ----------
__global__ void k_ct1(const float* __restrict__ shp, const float* __restrict__ w,
                      const float* __restrict__ bias, float* __restrict__ h) {
    int idx = blockIdx.x * 256 + threadIdx.x;
    const int total = 2 * 18 * H_H * H_H;
    if (idx >= total) return;
    int x = idx % H_H; int t = idx / H_H;
    int y = t % H_H;   t /= H_H;
    int oc = t % 18;   int b = t / 18;
    float acc = bias[oc];
    for (int ic = 0; ic < 4; ++ic)
        for (int i = 0; i < 4; ++i) {
            int yy = y - i;
            if (yy < 0 || yy >= 123) continue;
            for (int j = 0; j < 4; ++j) {
                int xx = x - j;
                if (xx < 0 || xx >= 123) continue;
                acc += shp[((b * 4 + ic) * 123 + yy) * 123 + xx] *
                       w[((ic * 18 + oc) * 4 + i) * 4 + j];
            }
        }
    h[idx] = fmaxf(acc, 0.f);
}

// ---------- ConvTranspose2d(18->18, k=3, s=1) + ReLU ----------
__global__ void k_ct2(const float* __restrict__ h, const float* __restrict__ w,
                      const float* __restrict__ bias, float* __restrict__ off) {
    int idx = blockIdx.x * 256 + threadIdx.x;
    const int total = 2 * 18 * OFF_HW * OFF_HW;
    if (idx >= total) return;
    int x = idx & 127; int t = idx >> 7;
    int y = t & 127;   t >>= 7;
    int oc = t % 18;   int b = t / 18;
    float acc = bias[oc];
    for (int ic = 0; ic < 18; ++ic)
        for (int i = 0; i < 3; ++i) {
            int yy = y - i;
            if (yy < 0 || yy >= H_H) continue;
            for (int j = 0; j < 3; ++j) {
                int xx = x - j;
                if (xx < 0 || xx >= H_H) continue;
                acc += h[((b * 18 + ic) * H_H + yy) * H_H + xx] *
                       w[((ic * 18 + oc) * 3 + i) * 3 + j];
            }
        }
    off[idx] = fmaxf(acc, 0.f);
}

// ---------- W[o][c][n] (f32) -> Wbf[o][n*256 + c] (bf16) ----------
__global__ void k_wcvt(const float* __restrict__ w, __bf16* __restrict__ wbf) {
    int idx = blockIdx.x * 256 + threadIdx.x;
    const int total = COUT * KK;
    if (idx >= total) return;
    int o = idx / KK, r = idx % KK;
    int c = r / 9, n = r % 9;
    wbf[(size_t)o * KK + n * CIN + c] = (__bf16)w[idx];
}

// ---------- TDM: DMA one 256x32 bf16 weight tile into LDS (padded rows) ----
// D# per CDNA5 ISA. 2D tile, groups 2/3 zero.
// 6-arg builtin form (clang-23 / therock-10.0 headers): extra int32x8 + cpol.
__device__ __forceinline__ void tdm_load_A(const __bf16* gsrc, u32 lds_off) {
    unsigned long long ga = (unsigned long long)(uintptr_t)gsrc;
    v4u g0;
    g0[0] = 1u;                                   // count=1, user descriptor
    g0[1] = lds_off;                              // lds_addr (bytes)
    g0[2] = (u32)ga;                              // global_addr[31:0]
    g0[3] = (u32)((ga >> 32) & 0x1FFFFFFu)        // global_addr[56:32]
            | 0x80000000u;                        // type = 2 ("image")
    v8i g1;
    g1[0] = (int)((1u << 16)                      // data_size = 2 bytes
            | (1u << 20)                          // pad_enable
            | (3u << 22)                          // pad_interval: 16 DWORDs (64B row)
            | (3u << 25));                        // pad_amount: 4 DWORDs (16B pad)
    g1[1] = (int)(32u << 16);                     // tensor_dim0 = 32 (bits 63:48)
    g1[2] = (int)(256u << 16);                    // tensor_dim1 = 256 (bits 95:80)
    g1[3] = (int)(32u << 16);                     // tile_dim0 = 32 (bits 127:112)
    g1[4] = 256;                                  // tile_dim1 = 256 (bits 143:128)
    g1[5] = (int)KK;                              // tensor_dim0_stride = 2304
    g1[6] = 0;
    g1[7] = 0;
    v4i g2 = {0, 0, 0, 0};
    v4i g3 = {0, 0, 0, 0};
    v8i g4 = {0, 0, 0, 0, 0, 0, 0, 0};
    __builtin_amdgcn_tensor_load_to_lds(g0, g1, g2, g3, g4, 0);
}

// ---------- Fused deformable im2col + bf16 WMMA GEMM ----------
// grid: (256 pixel-tiles of 64, 2 batches), block: 256 threads (8 wave32)
// Per block: 256 (M) x 64 (N) f32 output, K = 2304 in 72 steps of 32.
// A-tile is streamed by the Tensor Data Mover into a double buffer.
__global__ void __launch_bounds__(256)
k_deform_wmma(const float* __restrict__ x, const float* __restrict__ off,
              const __bf16* __restrict__ wbf, float* __restrict__ out) {
    __shared__ __align__(16) __bf16 sA[2][COUT * ASTR]; // double-buffered A tile
    __shared__ __align__(16) __bf16 sB[PIX * BSTR];     // 64(pix) x 32(k) B tile
    __shared__ float4 sW4[9 * PIX];                     // bilinear weights
    __shared__ int4   sI4[9 * PIX];                     // gather element indices

    const int b  = blockIdx.y;
    const int p0 = blockIdx.x * PIX;
    const int tid  = threadIdx.x;
    const int wv   = tid >> 5;          // wave 0..7
    const int lane = tid & 31;
    const int hh   = lane >> 4;         // half-wave (0/1)
    const int l    = lane & 15;

    // kick off the TDM for K-step 0 while we compute bilinear metadata
    if (tid < 32)
        tdm_load_A(wbf, (u32)(uintptr_t)&sA[0][0]);

    // ---- stage 0: per-(pixel, n) bilinear metadata ----
    for (int e = tid; e < 9 * PIX; e += 256) {
        int n = e >> 6, p = e & 63;
        int pg = p0 + p;
        int ho = pg >> 7, wo = pg & 127;
        float offy = off[(((size_t)(b * 18 + n)) << 14) + pg];
        float offx = off[(((size_t)(b * 18 + n + 9)) << 14) + pg];
        float py = offy + (float)((n / 3) - 1 + ho + 1);
        float px = offx + (float)((n % 3) - 1 + wo + 1);
        float fy = floorf(py), fx = floorf(px);
        int ity = (int)fy, itx = (int)fx;
        int lty = clampi(ity, 0, 127),     ltx = clampi(itx, 0, 127);
        int rby = clampi(ity + 1, 0, 127), rbx = clampi(itx + 1, 0, 127);
        float pyc = fminf(fmaxf(py, 0.f), 127.f);
        float pxc = fminf(fmaxf(px, 0.f), 127.f);
        float gy_lt = 1.f + ((float)lty - pyc);
        float gy_rb = 1.f - ((float)rby - pyc);
        float gx_lt = 1.f + ((float)ltx - pxc);
        float gx_rb = 1.f - ((float)rbx - pxc);
        sW4[e] = make_float4(gy_lt * gx_lt, gy_rb * gx_rb, gy_lt * gx_rb, gy_rb * gx_lt);
        sI4[e] = make_int4(lty * 128 + ltx, rby * 128 + rbx,
                           lty * 128 + rbx, rby * 128 + ltx);
    }
    __syncthreads();

    v8f acc[2][4] = {};
    const float* xb = x + ((size_t)b << 22);  // scalar base -> saddr-form gathers
    const int p   = tid & 63;                 // B-compute pixel (fixed per thread)
    const int cb  = (tid >> 6) * 8;           // B-compute channel group (8 consecutive)

    for (int n = 0; n < 9; ++n) {
        // per-n gather metadata (constant over the c-chunk loop)
        const float4 g  = sW4[n * PIX + p];
        const int4   iv = sI4[n * PIX + p];

        for (int cc = 0; cc < 8; ++cc) {
            const int c0 = cc * 32;
            const int ks = n * 8 + cc;        // flat K-step 0..71
            const __bf16* sAc = sA[ks & 1];
            __syncthreads();   // prior fragment reads done; sB free to overwrite

            // ---- compute B tile: 8 consecutive channels of one pixel row ----
            {
                v8bf pack;
                #pragma unroll
                for (int j = 0; j < 8; ++j) {
                    u32 ch = (u32)((c0 + cb + j) << 14);   // channel plane offset
                    float v = g.x * xb[ch + (u32)iv.x] + g.y * xb[ch + (u32)iv.y] +
                              g.z * xb[ch + (u32)iv.z] + g.w * xb[ch + (u32)iv.w];
                    pack[j] = (__bf16)v;
                }
                *(v8bf*)(sB + p * BSTR + cb) = pack;       // one ds_store_b128
            }
            // ---- TDM: prefetch next A tile, ensure current one landed ----
            if (tid < 32) {
                if (ks < 71) {
                    int ks2 = ks + 1;
                    int goff = ((ks2 >> 3) << 8) + ((ks2 & 7) << 5); // n2*256 + c2
                    tdm_load_A(wbf + goff, (u32)(uintptr_t)&sA[ks2 & 1][0]);
                    __builtin_amdgcn_s_wait_tensorcnt(1);  // current tile complete
                } else {
                    __builtin_amdgcn_s_wait_tensorcnt(0);
                }
            }
            __syncthreads();   // tiles visible to all waves

            // ---- WMMA: each wave does 2 M-tiles x 4 N-tiles ----
            v16bf afr[2], bfr[4];
            #pragma unroll
            for (int q = 0; q < 2; ++q) {
                const u32* arow = (const u32*)(sAc + (wv * 32 + q * 16 + l) * ASTR);
                v8u a;
                #pragma unroll
                for (int j = 0; j < 4; ++j) {
                    a[j]     = arow[hh * 4 + j];        // K = hh*8 + 2j(+1)
                    a[4 + j] = arow[8 + hh * 4 + j];    // K = 16 + hh*8 + 2j(+1)
                }
                afr[q] = __builtin_bit_cast(v16bf, a);
            }
            #pragma unroll
            for (int r = 0; r < 4; ++r) {
                const u32* brow = (const u32*)(sB + (r * 16 + l) * BSTR);
                v8u bb;
                #pragma unroll
                for (int j = 0; j < 8; ++j)
                    bb[j] = brow[(j >> 2) * 8 + hh * 4 + (j & 3)]; // K = (j>>2)*16 + hh*8 + 2(j&3)(+1), N = l
                bfr[r] = __builtin_bit_cast(v16bf, bb);
            }
            #pragma unroll
            for (int q = 0; q < 2; ++q)
                #pragma unroll
                for (int r = 0; r < 4; ++r)
                    acc[q][r] = __builtin_amdgcn_wmma_f32_16x16x32_bf16(
                        false, afr[q], false, bfr[r], (short)0, acc[q][r],
                        false, false);
        }
    }

    // ---- epilogue: D layout -> out[b][o][pg] ----
    #pragma unroll
    for (int q = 0; q < 2; ++q)
        #pragma unroll
        for (int r = 0; r < 4; ++r) {
            int pg = p0 + r * 16 + l;
            #pragma unroll
            for (int rr = 0; rr < 8; ++rr) {
                int o = wv * 32 + q * 16 + hh * 8 + rr;
                out[(((size_t)(b * COUT + o)) << 14) + pg] = acc[q][r][rr];
            }
        }
}

extern "C" void kernel_launch(void* const* d_in, const int* in_sizes, int n_in,
                              void* d_out, int out_size, void* d_ws, size_t ws_size,
                              hipStream_t stream) {
    const float* x      = (const float*)d_in[0];   // (2,256,128,128)
    const float* shape  = (const float*)d_in[1];   // (2,4,123,123)
    const float* ct1_w  = (const float*)d_in[2];   // (4,18,4,4)
    const float* ct1_b  = (const float*)d_in[3];   // (18,)
    const float* ct2_w  = (const float*)d_in[4];   // (18,18,3,3)
    const float* ct2_b  = (const float*)d_in[5];   // (18,)
    const float* dconvw = (const float*)d_in[6];   // (256,256,3,3)
    float* out = (float*)d_out;

    // workspace layout
    float*  hbuf = (float*)d_ws;                       // 2*18*126*126
    float*  offb = hbuf + 2 * 18 * H_H * H_H;          // 2*18*128*128
    __bf16* wbf  = (__bf16*)(offb + 2 * 18 * OFF_HW * OFF_HW); // 256*2304 bf16

    {
        int n1 = 2 * 18 * H_H * H_H;
        k_ct1<<<(n1 + 255) / 256, 256, 0, stream>>>(shape, ct1_w, ct1_b, hbuf);
    }
    {
        int n2 = 2 * 18 * OFF_HW * OFF_HW;
        k_ct2<<<(n2 + 255) / 256, 256, 0, stream>>>(hbuf, ct2_w, ct2_b, offb);
    }
    {
        int n3 = COUT * KK;
        k_wcvt<<<(n3 + 255) / 256, 256, 0, stream>>>(dconvw, wbf);
    }
    {
        dim3 grid(OFF_HW * OFF_HW / PIX, 2, 1);   // 256 pixel tiles, 2 batches
        k_deform_wmma<<<grid, 256, 0, stream>>>(x, offb, wbf, out);
    }
}
